// LSTM_6820408066677
// MI455X (gfx1250) — matrix operand order repaired and checked by reading, not compiled, not run on hardware
//
#include <hip/hip_runtime.h>

typedef __attribute__((ext_vector_type(16))) __bf16 v16bf;
typedef __attribute__((ext_vector_type(8)))  __bf16 v8bf;
typedef __attribute__((ext_vector_type(8)))  float  v8f;

constexpr int kB = 512;   // batch
constexpr int kS = 1024;  // sequence length
constexpr int kH = 128;   // hidden
constexpr int kC = 10;    // classes
constexpr int kBM = 16;   // batch rows per workgroup
constexpr int kThreads = 512;          // 16 waves of 32
constexpr int ZSTRIDE = 516;           // padded z row stride (dwords)
constexpr int HSTRIDE = 136;           // padded h row stride (bf16): 272B = 17*16B, +4 banks/row

__device__ __forceinline__ float sigmoid_f(float x) {
    return 1.0f / (1.0f + __expf(-x));
}
__device__ __forceinline__ float tanh_f(float x) {
    float e = __expf(2.0f * x);
    return 1.0f - 2.0f / (e + 1.0f);
}

__global__ void __launch_bounds__(kThreads, 1)
lstm_persistent(const float* __restrict__ x,
                const float* __restrict__ Wgx, const float* __restrict__ Wgh, const float* __restrict__ bg,
                const float* __restrict__ Wix, const float* __restrict__ Wih, const float* __restrict__ bi,
                const float* __restrict__ Wfx, const float* __restrict__ Wfh, const float* __restrict__ bfb,
                const float* __restrict__ Wox, const float* __restrict__ Woh, const float* __restrict__ bo,
                const float* __restrict__ Wph, const float* __restrict__ bp,
                float* __restrict__ out)
{
    __shared__ float  sh_z[kBM * ZSTRIDE];   // raw GEMM output z = h @ Wh, [16][512] padded
    __shared__ __bf16 sh_h[kBM * HSTRIDE];   // h (bf16) for WMMA A operand, [16][128] padded
    __shared__ float  sh_xt[kBM];            // x[:, t] for current step

    const int tid  = threadIdx.x;
    const int lane = tid & 31;
    const int w    = tid >> 5;               // wave id 0..15
    const int bm0  = blockIdx.x * kBM;

    // ---------------- loop-invariant B fragments (Wh slice, bf16, in VGPRs) ----------------
    // wave w owns 32 concatenated-gate columns [w*32, w*32+32); 32 | 128 -> single gate matrix
    const int gate = w >> 2;
    const float* Wmat = (gate == 0) ? Wgh : (gate == 1) ? Wih : (gate == 2) ? Wfh : Woh;
    const int nloc = (w & 3) * 32;
    const int lx = lane & 15;
    const int khalf16 = (lane & 16) ? 16 : 0;  // this code's A/B k order: lanes 0-15 hold K=0..15, lanes 16-31 K=16..31

    v16bf bfrag[2][4];
    #pragma unroll
    for (int tile = 0; tile < 2; ++tile) {
        const int col = nloc + tile * 16 + lx;
        #pragma unroll
        for (int s = 0; s < 4; ++s) {
            const int kb = 32 * s + khalf16;
            v16bf bb;
            #pragma unroll
            for (int e = 0; e < 16; ++e)
                bb[e] = (__bf16)Wmat[(kb + e) * kH + col];
            bfrag[tile][s] = bb;
        }
    }

    // ---------------- gate-phase loop invariants ----------------
    const int hcol = tid & 127;
    const int mq   = tid >> 7;               // 0..3 ; thread owns rows m = mq*4 + p
    const float wx_g = Wgx[hcol], wx_i = Wix[hcol], wx_f = Wfx[hcol], wx_o = Wox[hcol];
    float bias_g[4], bias_i[4], bias_f[4], bias_o[4];
    #pragma unroll
    for (int p = 0; p < 4; ++p) {
        const int m = mq * 4 + p;
        const int gidx = (bm0 + m) * kH + hcol;
        bias_g[p] = bg[gidx];  bias_i[p] = bi[gidx];
        bias_f[p] = bfb[gidx]; bias_o[p] = bo[gidx];
    }
    float creg[4] = {0.f, 0.f, 0.f, 0.f};
    float hreg[4] = {0.f, 0.f, 0.f, 0.f};

    // zero initial h
    for (int idx = tid; idx < kBM * HSTRIDE; idx += kThreads) sh_h[idx] = (__bf16)0.0f;
    __syncthreads();

    const int ncol0  = w * 32;
    const int khalf8 = (lane & 16) ? 8 : 0;  // A layout half-wave K offset
    const int rowoff = (lane & 16) ? 8 : 0;  // C/D layout half-wave M offset

    // ---------------- persistent recurrence over S steps ----------------
    for (int t = 0; t < kS; ++t) {
        if (tid < kBM) sh_xt[tid] = x[(bm0 + tid) * kS + t];  // overlaps with GEMM

        // A fragments of h[16][128] (bf16): lane m=lx; K chunks [kb..kb+7] and [kb+8..kb+15]
        v16bf afrag[4];
        #pragma unroll
        for (int s = 0; s < 4; ++s) {
            const int base = lx * HSTRIDE + s * 32 + khalf16;
            v8bf lo = *(const v8bf*)&sh_h[base];
            v8bf hi = *(const v8bf*)&sh_h[base + 8];
            afrag[s] = __builtin_shufflevector(lo, hi,
                        0,1,2,3,4,5,6,7,8,9,10,11,12,13,14,15);
        }

        v8f c0 = {}; v8f c1 = {};
        #pragma unroll
        for (int s = 0; s < 4; ++s) {
            c0 = __builtin_amdgcn_wmma_f32_16x16x32_bf16(false, afrag[s], false, bfrag[0][s],
                                                         (short)0, c0, false, false);
            c1 = __builtin_amdgcn_wmma_f32_16x16x32_bf16(false, afrag[s], false, bfrag[1][s],
                                                         (short)0, c1, false, false);
        }
        #pragma unroll
        for (int r = 0; r < 8; ++r) {
            const int row = rowoff + r;
            sh_z[row * ZSTRIDE + ncol0 + lx]      = c0[r];
            sh_z[row * ZSTRIDE + ncol0 + 16 + lx] = c1[r];
        }
        __syncthreads();

        // gate math: z += x_t * Wx + bias ; LSTM cell update
        #pragma unroll
        for (int p = 0; p < 4; ++p) {
            const int m = mq * 4 + p;
            const float xv = sh_xt[m];
            const float zg = sh_z[m * ZSTRIDE + hcol]       + xv * wx_g + bias_g[p];
            const float zi = sh_z[m * ZSTRIDE + 128 + hcol] + xv * wx_i + bias_i[p];
            const float zf = sh_z[m * ZSTRIDE + 256 + hcol] + xv * wx_f + bias_f[p];
            const float zo = sh_z[m * ZSTRIDE + 384 + hcol] + xv * wx_o + bias_o[p];
            const float g  = tanh_f(zg);
            const float ii = sigmoid_f(zi);
            const float ff = sigmoid_f(zf);
            const float oo = sigmoid_f(zo);
            const float c  = g * ii + creg[p] * ff;
            const float h  = tanh_f(c) * oo;
            creg[p] = c; hreg[p] = h;
            sh_h[m * HSTRIDE + hcol] = (__bf16)h;
        }
        __syncthreads();
    }

    // ---------------- final projection: out = h @ Wph + bp ----------------
    #pragma unroll
    for (int p = 0; p < 4; ++p) {
        const int m = mq * 4 + p;
        sh_z[m * kH + hcol] = hreg[p];   // reuse z buffer for f32 h
    }
    __syncthreads();
    if (tid < kBM * kC) {
        const int m = tid / kC, cc = tid % kC;
        float acc = bp[(bm0 + m) * kC + cc];
        #pragma unroll 8
        for (int k = 0; k < kH; ++k)
            acc += sh_z[m * kH + k] * Wph[k * kC + cc];
        out[(bm0 + m) * kC + cc] = acc;
    }
}

extern "C" void kernel_launch(void* const* d_in, const int* in_sizes, int n_in,
                              void* d_out, int out_size, void* d_ws, size_t ws_size,
                              hipStream_t stream) {
    const float* x   = (const float*)d_in[0];
    const float* Wgx = (const float*)d_in[1];
    const float* Wgh = (const float*)d_in[2];
    const float* bg  = (const float*)d_in[3];
    const float* Wix = (const float*)d_in[4];
    const float* Wih = (const float*)d_in[5];
    const float* bi  = (const float*)d_in[6];
    const float* Wfx = (const float*)d_in[7];
    const float* Wfh = (const float*)d_in[8];
    const float* bfb = (const float*)d_in[9];
    const float* Wox = (const float*)d_in[10];
    const float* Woh = (const float*)d_in[11];
    const float* bo  = (const float*)d_in[12];
    const float* Wph = (const float*)d_in[13];
    const float* bp  = (const float*)d_in[14];
    float* out = (float*)d_out;

    lstm_persistent<<<kB / kBM, kThreads, 0, stream>>>(
        x, Wgx, Wgh, bg, Wix, Wih, bi, Wfx, Wfh, bfb, Wox, Woh, bo, Wph, bp, out);
}
